// Text2Motion_Transformer_54296976556356
// MI455X (gfx1250) — matrix operand order repaired and checked by reading, not compiled, not run on hardware
//
#include <hip/hip_runtime.h>

// ---------------------------------------------------------------------------
// Text2Motion transformer forward, CDNA5 (gfx1250, wave32, WMMA) version.
// All GEMMs run on v_wmma_f32_16x16x32_bf16 with fp32 accumulation.
// Residual stream, layernorm stats and softmax are kept in fp32.
// Requires ws_size >= ~520 MB (bf16 weights + activation scratch).
// ---------------------------------------------------------------------------

typedef __attribute__((ext_vector_type(16))) __bf16 v16bf;
typedef __attribute__((ext_vector_type(8)))  float  v8f;

#define LDSA 72    // LDS row stride for 64-wide K slab (64 + 8 pad), bf16 elems

// --- WMMA fragment gathers (ISA 7.12.2 per-lane layouts) --------------------
// A (16x32 bf16): lane -> M = lane%16 ; elems 0..7 -> K = base0..base0+7,
// elems 8..15 -> K = base0+16..base0+23 ; base0 = 8*(lane>=16).
// Both 8-elem chunks are 16B-contiguous in LDS -> two ds_load_b128.
__device__ __forceinline__ v16bf frag_a(const __bf16* s, int row0, int kofs, int lane) {
  const __bf16* p = s + (size_t)(row0 + (lane & 15)) * LDSA + kofs + ((lane >> 4) << 3);
  v16bf f;
  __builtin_memcpy(&f, p, 16);
  __builtin_memcpy((char*)&f + 16, p + 16, 16);
  return f;
}
// B (32x16 bf16): lane -> N = lane%16 ; same K layout as A: elems 0..7 -> K = base0..base0+7,
// elems 8..15 -> K = base0+16..base0+23 ; base0 = 8*(lane>=16) -> two 16B LDS reads.
__device__ __forceinline__ v16bf frag_b(const __bf16* s, int col0, int kofs, int lane) {
  const __bf16* p = s + (size_t)(col0 + (lane & 15)) * LDSA + kofs + ((lane >> 4) << 3);
  v16bf f;
  __builtin_memcpy(&f, p, 16);
  __builtin_memcpy((char*)&f + 16, p + 16, 16);
  return f;
}

// --- Generic batched WMMA GEMM ---------------------------------------------
// C[M,N] = alpha * A[M,K] @ B^T (+bias) (+resid), B given as [N][K] row-major.
// Batch z decomposed as (zb, zh) = (z/Hb, z%Hb) with independent pointer
// strides for A, B, C.
// mode 0: fp32 out (+resid). mode 1: bf16 out. mode 2: bf16 gelu(out).
// Block: 256 threads = 8 waves; tile 128x128; K slab 64 (2 WMMA k-steps).
// Each wave owns 32x64 (2x4 accumulators) -> 16 WMMAs per barrier pair.
__global__ __launch_bounds__(256)
void gemm_wmma_bf16(const __bf16* __restrict__ A, long long lda, long long sAb, long long sAh,
                    const __bf16* __restrict__ Bw, long long ldb, long long sBb, long long sBh,
                    float* __restrict__ Cf, __bf16* __restrict__ Ch,
                    long long ldc, long long sCb, long long sCh,
                    const float* __restrict__ bias, const float* __restrict__ resid,
                    int M, int N, int K, int Hb, float alpha, int mode)
{
  __shared__ __bf16 As[128 * LDSA];
  __shared__ __bf16 Bs[128 * LDSA];

  int z  = blockIdx.z;
  int zb = z / Hb, zh = z % Hb;
  A  += (size_t)zb * sAb + (size_t)zh * sAh;
  Bw += (size_t)zb * sBb + (size_t)zh * sBh;
  long long coff = (long long)zb * sCb + (long long)zh * sCh;

  int rowBase = blockIdx.y * 128;
  int colBase = blockIdx.x * 128;
  int t = threadIdx.x;
  int lane = t & 31, wave = t >> 5;
  int waveM = wave & 3;   // 4 row groups of 32 rows
  int waveN = wave >> 2;  // 2 col groups of 64 cols

  int r    = t >> 1, half = t & 1;
  int gnB  = colBase + r;
  const __bf16* gA = A + (size_t)(rowBase + r) * lda + half * 32;
  const __bf16* gB = Bw + (size_t)gnB * ldb + half * 32;
  uint4* sA = (uint4*)(As + r * LDSA + half * 32);
  uint4* sB = (uint4*)(Bs + r * LDSA + half * 32);

  v8f acc[8] = {};

  for (int k0 = 0; k0 < K; k0 += 64) {
    // ---- stage A+B tiles: gather ALL global loads into registers first so
    // the backend can clause them, then drain into LDS with ds_store_b128 ----
    uint4 a0, a1, a2, a3, b0, b1, b2, b3;
    {
      const uint4* g = (const uint4*)(gA + k0);
      a0 = g[0]; a1 = g[1]; a2 = g[2]; a3 = g[3];
      if (k0 + 64 < K) __builtin_prefetch(gA + k0 + 64, 0, 1);  // global_prefetch_b8
    }
    if (gnB < N) {
      const uint4* g = (const uint4*)(gB + k0);
      b0 = g[0]; b1 = g[1]; b2 = g[2]; b3 = g[3];
      if (k0 + 64 < K) __builtin_prefetch(gB + k0 + 64, 0, 1);
    } else {
      uint4 zz = make_uint4(0u, 0u, 0u, 0u);
      b0 = zz; b1 = zz; b2 = zz; b3 = zz;
    }
    sA[0] = a0; sA[1] = a1; sA[2] = a2; sA[3] = a3;
    sB[0] = b0; sB[1] = b1; sB[2] = b2; sB[3] = b3;
    __syncthreads();

    // ---- 16 WMMAs per K slab: 2 k-steps x (2 A-frags x 4 B-frags) ----
#pragma unroll
    for (int ks = 0; ks < 2; ++ks) {
      v16bf af0 = frag_a(As, waveM * 32,      ks * 32, lane);
      v16bf af1 = frag_a(As, waveM * 32 + 16, ks * 32, lane);
#pragma unroll
      for (int nt = 0; nt < 4; ++nt) {
        v16bf bf = frag_b(Bs, waveN * 64 + nt * 16, ks * 32, lane);
        acc[nt]     = __builtin_amdgcn_wmma_f32_16x16x32_bf16(
                          false, af0, false, bf, (short)0, acc[nt],     false, false);
        acc[4 + nt] = __builtin_amdgcn_wmma_f32_16x16x32_bf16(
                          false, af1, false, bf, (short)0, acc[4 + nt], false, false);
      }
    }
    __syncthreads();
  }

  // ---- epilogue (C/D layout: VGPR r, lanes<16 -> M=r, lanes>=16 -> M=r+8) ----
  int rowHalf = (lane >> 4) * 8;
  int colIn   = lane & 15;
#pragma unroll
  for (int mt = 0; mt < 2; ++mt)
#pragma unroll
    for (int nt = 0; nt < 4; ++nt)
#pragma unroll
      for (int rr = 0; rr < 8; ++rr) {
        int gm = rowBase + waveM * 32 + mt * 16 + rowHalf + rr;
        int gn = colBase + waveN * 64 + nt * 16 + colIn;
        if (gm < M && gn < N) {
          float v = acc[mt * 4 + nt][rr] * alpha;
          if (bias) v += bias[gn];
          long long ci = coff + (long long)gm * ldc + gn;
          if (mode == 0) {
            if (resid) v += resid[ci];
            Cf[ci] = v;
          } else if (mode == 1) {
            Ch[ci] = (__bf16)v;
          } else {
            float g = 0.5f * v * (1.0f + erff(v * 0.70710678118654752f));
            Ch[ci] = (__bf16)g;
          }
        }
      }
}

// --- fp32 -> bf16 conversion (weights) -------------------------------------
__global__ void f32_to_bf16(const float* __restrict__ s, __bf16* __restrict__ d, long long n) {
  long long i  = (long long)blockIdx.x * blockDim.x + threadIdx.x;
  long long st = (long long)gridDim.x * blockDim.x;
  for (; i < n; i += st) d[i] = (__bf16)s[i];
}

// --- V transpose: [b*T+t][h*HD+d] -> [(b*H+h)*HD+d][t] (bf16) --------------
// Makes the AV GEMM use the standard [N][K] B-operand path.
__global__ void transpose_v(const __bf16* __restrict__ v, __bf16* __restrict__ vt,
                            int B_, int T_, int D_, int H_, int HD_)
{
  long long n  = (long long)B_ * H_ * HD_ * T_;
  long long i  = (long long)blockIdx.x * blockDim.x + threadIdx.x;
  long long st = (long long)gridDim.x * blockDim.x;
  for (; i < n; i += st) {
    int tt = (int)(i % T_);
    long long r = i / T_;            // (b*H + h)*HD + d
    int d  = (int)(r % HD_);
    long long bh = r / HD_;
    int h  = (int)(bh % H_);
    int b  = (int)(bh / H_);
    vt[i] = v[((size_t)b * T_ + tt) * D_ + (size_t)h * HD_ + d];
  }
}

// --- embedding + clip conditioning + sinusoidal PE -------------------------
__global__ __launch_bounds__(256)
void embed_kernel(const int* __restrict__ idx, const float* __restrict__ clip,
                  const float* __restrict__ tok_emb,
                  const float* __restrict__ wc1, const float* __restrict__ bc1,
                  const float* __restrict__ wc2, const float* __restrict__ bc2,
                  float* __restrict__ x, int T, int D, int CLIPN)
{
  int tok = blockIdx.x;
  int b = tok / T, t = tok % T;
  int id = idx[tok];
  for (int d = threadIdx.x; d < D; d += blockDim.x) {
    float val;
    if (id == -1 || id == -2) {
      const float* w  = (id == -1) ? wc1 : wc2;
      const float* bb = (id == -1) ? bc1 : bc2;
      float s = 0.f;
      for (int k = 0; k < CLIPN; ++k)
        s += clip[(size_t)b * CLIPN + k] * w[(size_t)d * CLIPN + k];
      val = s + bb[d];
    } else {
      val = tok_emb[(size_t)id * D + d];
    }
    int p = d >> 1;
    float ang = (float)t * __powf(10000.0f, -(2.0f * p) / (float)D);
    val += (d & 1) ? __cosf(ang) : __sinf(ang);
    x[(size_t)tok * D + d] = val;
  }
}

// --- layernorm: fp32 in -> bf16 out (feeds WMMA GEMMs) ---------------------
__global__ __launch_bounds__(256)
void layernorm_bf16(const float* __restrict__ x, const float* __restrict__ w,
                    const float* __restrict__ b, __bf16* __restrict__ out, int D)
{
  int row = blockIdx.x;
  const float* xr = x + (size_t)row * D;
  __shared__ float rs[256], rq[256];
  float s = 0.f, q = 0.f;
  for (int d = threadIdx.x; d < D; d += 256) { float v = xr[d]; s += v; q += v * v; }
  rs[threadIdx.x] = s; rq[threadIdx.x] = q;
  __syncthreads();
  for (int o = 128; o > 0; o >>= 1) {
    if (threadIdx.x < o) {
      rs[threadIdx.x] += rs[threadIdx.x + o];
      rq[threadIdx.x] += rq[threadIdx.x + o];
    }
    __syncthreads();
  }
  float mu  = rs[0] / D;
  float var = rq[0] / D - mu * mu;
  float inv = rsqrtf(var + 1e-5f);
  for (int d = threadIdx.x; d < D; d += 256)
    out[(size_t)row * D + d] = (__bf16)((xr[d] - mu) * inv * w[d] + b[d]);
}

// --- masked softmax (reference two-triangle mask), fp32 -> bf16 probs ------
__device__ __forceinline__ bool attn_ok(int i, int j, int m, int T) {
  bool left  = (j <= i) && (i < m) && (j <= m - 1 - i);
  bool tr    = (i < m) ? (j >= m - 1 - i) : (j >= T - 1 - i);
  bool right = tr && (j >= i);
  return (((left || right) && (j < m)) || (j == 0));
}

__global__ __launch_bounds__(256)
void softmax_mask(const float* __restrict__ att, __bf16* __restrict__ p,
                  const int* __restrict__ mlen, int T, int Hh)
{
  long long row = blockIdx.x;                 // ((b*H)+h)*T + i
  int i = (int)(row % T);
  int b = (int)(row / ((long long)T * Hh));
  int m = mlen[b];
  const float* a = att + row * T;
  __bf16* o = p + row * T;
  __shared__ float red[256];

  float vals[2];
  float mx = -1e30f;
#pragma unroll
  for (int q = 0; q < 2; ++q) {               // T == 512 == 2*256
    int c = q * 256 + threadIdx.x;
    float v = attn_ok(i, c, m, T) ? a[c] : -1e30f;
    vals[q] = v;
    mx = fmaxf(mx, v);
  }
  red[threadIdx.x] = mx; __syncthreads();
  for (int s = 128; s > 0; s >>= 1) {
    if (threadIdx.x < s) red[threadIdx.x] = fmaxf(red[threadIdx.x], red[threadIdx.x + s]);
    __syncthreads();
  }
  mx = red[0]; __syncthreads();

  float sum = 0.f;
#pragma unroll
  for (int q = 0; q < 2; ++q) { vals[q] = __expf(vals[q] - mx); sum += vals[q]; }
  red[threadIdx.x] = sum; __syncthreads();
  for (int s = 128; s > 0; s >>= 1) {
    if (threadIdx.x < s) red[threadIdx.x] += red[threadIdx.x + s];
    __syncthreads();
  }
  float inv = 1.0f / red[0];
#pragma unroll
  for (int q = 0; q < 2; ++q)
    o[q * 256 + threadIdx.x] = (__bf16)(vals[q] * inv);
}

// ---------------------------------------------------------------------------
extern "C" void kernel_launch(void* const* d_in, const int* in_sizes, int n_in,
                              void* d_out, int out_size, void* d_ws, size_t ws_size,
                              hipStream_t stream)
{
  (void)in_sizes; (void)n_in; (void)out_size; (void)ws_size;
  constexpr int Bc = 8, Tc = 512, Dc = 1024, Hc = 16, Lc = 8, Vc = 8192, CLIPc = 512, HDc = 64;
  constexpr int ROWS = Bc * Tc;               // 4096
  constexpr int FDc  = 4 * Dc;                // 4096

  const int*   idx     = (const int*)  d_in[0];
  const float* clip    = (const float*)d_in[1];
  const int*   mlen    = (const int*)  d_in[2];
  /* d_in[3] = max_len (compile-time Tc) */
  const float* tok_emb = (const float*)d_in[4];
  const float* w_cond  = (const float*)d_in[5];
  const float* b_cond  = (const float*)d_in[6];
  const float* w_cond2 = (const float*)d_in[7];
  const float* b_cond2 = (const float*)d_in[8];
  const float* ln1w = (const float*)d_in[9];
  const float* ln1b = (const float*)d_in[10];
  const float* wq = (const float*)d_in[11];
  const float* bq = (const float*)d_in[12];
  const float* wk = (const float*)d_in[13];
  const float* bk = (const float*)d_in[14];
  const float* wv = (const float*)d_in[15];
  const float* bv = (const float*)d_in[16];
  const float* wo = (const float*)d_in[17];
  const float* bo = (const float*)d_in[18];
  const float* ln2w = (const float*)d_in[19];
  const float* ln2b = (const float*)d_in[20];
  const float* w1 = (const float*)d_in[21];
  const float* b1 = (const float*)d_in[22];
  const float* w2 = (const float*)d_in[23];
  const float* b2 = (const float*)d_in[24];
  const float* lnfw = (const float*)d_in[25];
  const float* lnfb = (const float*)d_in[26];
  const float* whead = (const float*)d_in[27];

  // ---- workspace carve ----
  char* base = (char*)d_ws;
  size_t off = 0;
  auto carve = [&](size_t bytes) -> char* {
    off = (off + 255) & ~(size_t)255;
    char* p = base + off;
    off += bytes;
    return p;
  };
  __bf16* wq_b = (__bf16*)carve((size_t)Lc * Dc * Dc * 2);
  __bf16* wk_b = (__bf16*)carve((size_t)Lc * Dc * Dc * 2);
  __bf16* wv_b = (__bf16*)carve((size_t)Lc * Dc * Dc * 2);
  __bf16* wo_b = (__bf16*)carve((size_t)Lc * Dc * Dc * 2);
  __bf16* w1_b = (__bf16*)carve((size_t)Lc * FDc * Dc * 2);
  __bf16* w2_b = (__bf16*)carve((size_t)Lc * Dc * FDc * 2);
  __bf16* wh_b = (__bf16*)carve((size_t)Vc * Dc * 2);
  float*  xbuf = (float*) carve((size_t)ROWS * Dc * 4);
  __bf16* hbf  = (__bf16*)carve((size_t)ROWS * Dc * 2);
  __bf16* qb   = (__bf16*)carve((size_t)ROWS * Dc * 2);
  __bf16* kb   = (__bf16*)carve((size_t)ROWS * Dc * 2);
  __bf16* vb   = (__bf16*)carve((size_t)ROWS * Dc * 2);
  __bf16* vtb  = (__bf16*)carve((size_t)ROWS * Dc * 2);   // V transposed per head
  float*  att  = (float*) carve((size_t)Bc * Hc * Tc * Tc * 4);
  __bf16* pb   = (__bf16*)carve((size_t)Bc * Hc * Tc * Tc * 2);
  __bf16* yb   = (__bf16*)carve((size_t)ROWS * Dc * 2);
  __bf16* h1b  = (__bf16*)carve((size_t)ROWS * FDc * 2);

  auto conv = [&](const float* s, __bf16* d, long long n) {
    int grid = (int)((n + 256 * 8 - 1) / (256 * 8));
    f32_to_bf16<<<grid, 256, 0, stream>>>(s, d, n);
  };
  auto gemm = [&](const __bf16* Aa, long long lda, long long sAb, long long sAh,
                  const __bf16* Bb, long long ldb, long long sBb, long long sBh,
                  float* Cf, __bf16* Ch, long long ldc, long long sCb, long long sCh,
                  const float* bias, const float* resid,
                  int M, int N, int K, int Hb, int nbatch, float alpha, int mode) {
    dim3 g((N + 127) / 128, (M + 127) / 128, nbatch);
    gemm_wmma_bf16<<<g, 256, 0, stream>>>(Aa, lda, sAb, sAh, Bb, ldb, sBb, sBh,
                                          Cf, Ch, ldc, sCb, sCh, bias, resid,
                                          M, N, K, Hb, alpha, mode);
  };

  // ---- weights -> bf16 ----
  conv(wq, wq_b, (long long)Lc * Dc * Dc);
  conv(wk, wk_b, (long long)Lc * Dc * Dc);
  conv(wv, wv_b, (long long)Lc * Dc * Dc);
  conv(wo, wo_b, (long long)Lc * Dc * Dc);
  conv(w1, w1_b, (long long)Lc * FDc * Dc);
  conv(w2, w2_b, (long long)Lc * Dc * FDc);
  conv(whead, wh_b, (long long)Vc * Dc);

  // ---- embedding + PE ----
  embed_kernel<<<ROWS, 256, 0, stream>>>(idx, clip, tok_emb, w_cond, b_cond,
                                         w_cond2, b_cond2, xbuf, Tc, Dc, CLIPc);

  const long long TD  = (long long)Tc * Dc;
  const long long TT  = (long long)Tc * Tc;
  const long long HDT = (long long)HDc * Tc;

  for (int l = 0; l < Lc; ++l) {
    // LN1 -> bf16
    layernorm_bf16<<<ROWS, 256, 0, stream>>>(xbuf, ln1w + (size_t)l * Dc, ln1b + (size_t)l * Dc, hbf, Dc);
    // QKV projections (bf16 out + bias)
    gemm(hbf, Dc, 0, 0, wq_b + (size_t)l * Dc * Dc, Dc, 0, 0,
         nullptr, qb, Dc, 0, 0, bq + (size_t)l * Dc, nullptr, ROWS, Dc, Dc, 1, 1, 1.0f, 1);
    gemm(hbf, Dc, 0, 0, wk_b + (size_t)l * Dc * Dc, Dc, 0, 0,
         nullptr, kb, Dc, 0, 0, bk + (size_t)l * Dc, nullptr, ROWS, Dc, Dc, 1, 1, 1.0f, 1);
    gemm(hbf, Dc, 0, 0, wv_b + (size_t)l * Dc * Dc, Dc, 0, 0,
         nullptr, vb, Dc, 0, 0, bv + (size_t)l * Dc, nullptr, ROWS, Dc, Dc, 1, 1, 1.0f, 1);
    // V -> per-head transposed layout [ (b,h), d, t ]
    {
      long long nT = (long long)ROWS * Dc;
      int grid = (int)((nT + 256 * 8 - 1) / (256 * 8));
      transpose_v<<<grid, 256, 0, stream>>>(vb, vtb, Bc, Tc, Dc, Hc, HDc);
    }
    // scores: per (b,h) Q(512x64) @ K^T -> att fp32, scaled by 1/sqrt(64)
    gemm(qb, Dc, TD, HDc, kb, Dc, TD, HDc,
         att, nullptr, Tc, (long long)Hc * TT, TT, nullptr, nullptr,
         Tc, Tc, HDc, Hc, Bc * Hc, 0.125f, 0);
    // masked softmax -> bf16 probs
    softmax_mask<<<Bc * Hc * Tc, 256, 0, stream>>>(att, pb, mlen, Tc, Hc);
    // AV: P(512x512) @ V(512x64) via V^T as [N=64][K=512]; scatter to [rows,D]
    gemm(pb, Tc, (long long)Hc * TT, TT, vtb, Tc, (long long)Hc * HDT, HDT,
         nullptr, yb, Dc, TD, HDc, nullptr, nullptr,
         Tc, HDc, Tc, Hc, Bc * Hc, 1.0f, 1);
    // output projection + residual (fp32, in-place on x)
    gemm(yb, Dc, 0, 0, wo_b + (size_t)l * Dc * Dc, Dc, 0, 0,
         xbuf, nullptr, Dc, 0, 0, bo + (size_t)l * Dc, xbuf, ROWS, Dc, Dc, 1, 1, 1.0f, 0);
    // LN2 -> bf16
    layernorm_bf16<<<ROWS, 256, 0, stream>>>(xbuf, ln2w + (size_t)l * Dc, ln2b + (size_t)l * Dc, hbf, Dc);
    // MLP up + exact GELU -> bf16
    gemm(hbf, Dc, 0, 0, w1_b + (size_t)l * FDc * Dc, Dc, 0, 0,
         nullptr, h1b, FDc, 0, 0, b1 + (size_t)l * FDc, nullptr, ROWS, FDc, Dc, 1, 1, 1.0f, 2);
    // MLP down + residual (fp32, in-place on x)
    gemm(h1b, FDc, 0, 0, w2_b + (size_t)l * Dc * FDc, FDc, 0, 0,
         xbuf, nullptr, Dc, 0, 0, b2 + (size_t)l * Dc, xbuf, ROWS, Dc, FDc, 1, 1, 1.0f, 0);
  }

  // final LN + head GEMM -> logits (fp32, d_out)
  layernorm_bf16<<<ROWS, 256, 0, stream>>>(xbuf, lnfw, lnfb, hbf, Dc);
  gemm(hbf, Dc, 0, 0, wh_b, Dc, 0, 0,
       (float*)d_out, nullptr, Vc, 0, 0, nullptr, nullptr, ROWS, Vc, Dc, 1, 1, 1.0f, 0);
}
